// PrototypicalNetwork_29360396435773
// MI455X (gfx1250) — matrix operand order repaired and checked by reading, not compiled, not run on hardware
//
#include <hip/hip_runtime.h>
#include <hip/hip_bf16.h>

typedef __attribute__((ext_vector_type(16))) _Float16 v16h;
typedef __attribute__((ext_vector_type(8)))  _Float16 v8h;
typedef __attribute__((ext_vector_type(8)))  float    v8f;

#define NCLS 5
#define NGRP 4
#define CHN  64
#define DEMB 1600

// ---------------------------------------------------------------------------
// Pack conv weights (OIHW fp32) into the WMMA B-fragment layout for
// V_WMMA_F32_16X16X32_F16, with K reordered as k = (kh*3+kw)*Cin + c so a
// 32-wide K chunk has constant (kh,kw) when Cin==64.
// Layout: [chunk][ntile][lane][e] f16;  lane L holds N = L&15,
// element e holds K = chunk*32 + ((e>>3)<<4) + ((L>>4)<<3) + (e&7).
// ---------------------------------------------------------------------------
__global__ void pack_weights_k(const float* __restrict__ w, _Float16* __restrict__ wp,
                               int Cin, int nchunks) {
  int idx = blockIdx.x * blockDim.x + threadIdx.x;
  int total = nchunks * 4 * 32 * 16;
  if (idx >= total) return;
  int e     = idx & 15;
  int lane  = (idx >> 4) & 31;
  int ntile = (idx >> 9) & 3;
  int chunk = idx >> 11;
  int Ktot  = Cin * 9;
  int n = ntile * 16 + (lane & 15);
  int k = chunk * 32 + ((e >> 3) << 4) + ((lane >> 4) << 3) + (e & 7);
  float v = 0.f;
  if (k < Ktot) {
    int khkw = k / Cin;          // spatial index kh*3+kw
    int c    = k - khkw * Cin;   // input channel
    v = w[(size_t)n * Ktot + c * 9 + khkw];
  }
  wp[idx] = (_Float16)v;
}

// ---------------------------------------------------------------------------
// Implicit-GEMM 3x3 conv (pad 1, stride 1) + bias via WMMA f16->f32.
// Block = 128 threads = 4 wave32s. Wave w owns M-tile (blk*4+w): 16 pixels,
// and computes ALL 64 output channels (4 accumulators, 4 WMMAs per K-chunk).
// K order: k = (kh*3+kw)*CIN + c.
//  * CIN==64 path: input is NHWC f16 -> each chunk has constant (kh,kw);
//    the 16 A elements per lane are two CONTIGUOUS 8xf16 (16B) loads at one
//    pixel: channels cbase+kband+{0..7} and {16..23}.
//  * CIN==3 path (layer 1): generic NCHW fp32 gather, single chunk.
// Output: NCHW f32 (contiguous pixel stores per lane).
// Fragment layouts per CDNA5 ISA:
//   A: M = lane&15, Kloc = ((e>>3)<<4) + ((lane>>4)<<3) + (e&7)
//   B: N = lane&15, Kloc = ((e>>3)<<4) + ((lane>>4)<<3) + (e&7)
//   D: M = ((lane>>4)<<3)+r, N = lane&15
// ---------------------------------------------------------------------------
template <typename TIN, int CIN, int H, int W>
__global__ __launch_bounds__(128)
void conv_wmma_k(const TIN* __restrict__ in, const _Float16* __restrict__ wp,
                 const float* __restrict__ bias, float* __restrict__ out) {
  constexpr int HW      = H * W;
  constexpr int KTOT    = CIN * 9;
  constexpr int NCHUNKS = (KTOT + 31) / 32;
  constexpr int NMBLK   = (HW + 63) / 64;

  int lane  = threadIdx.x & 31;
  int wave  = threadIdx.x >> 5;
  int img   = blockIdx.x / NMBLK;
  int mblk  = blockIdx.x - img * NMBLK;
  int mtile = mblk * 4 + wave;

  const TIN* inp = in + (size_t)img * CIN * HW;

  int m = lane & 15;
  int p = mtile * 16 + m;
  bool pvalid = (p < HW);
  int h    = pvalid ? (p / W) : 0;
  int wcol = pvalid ? (p - h * W) : 0;
  int kband = (lane >> 4) << 3;

  v8f acc0 = {}, acc1 = {}, acc2 = {}, acc3 = {};

  for (int chunk = 0; chunk < NCHUNKS; ++chunk) {
    // ---- build A fragment (16 f16 per lane) ----
    v16h a;
    if constexpr (CIN == 64) {
      int khkw = chunk >> 1;             // constant (kh,kw) within chunk
      int kh = khkw / 3;
      int kw = khkw - kh * 3;
      int cbase = (chunk & 1) << 5;      // 0 or 32
      int ih = h + kh - 1;
      int iw = wcol + kw - 1;
      bool valid = pvalid && (ih >= 0) && (ih < H) && (iw >= 0) && (iw < W);
      // NHWC: all channels of one pixel are contiguous
      const TIN* bp = inp + (size_t)(ih * W + iw) * CHN + cbase + kband;
      v8h lo = {}, hi = {};
      if (valid) {
        lo = *(const v8h*)(bp);          // channels +0..7   (16B)
        hi = *(const v8h*)(bp + 16);     // channels +16..23 (16B)
      }
#pragma unroll
      for (int e = 0; e < 8; ++e) {
        a[e]     = lo[e];
        a[e + 8] = hi[e];
      }
    } else {
      // generic path (layer 1, CIN==3 NCHW fp32, single chunk)
#pragma unroll
      for (int e = 0; e < 16; ++e) {
        int kloc = ((e >> 3) << 4) + kband + (e & 7);
        int k = (chunk << 5) + kloc;
        float v = 0.f;
        if (pvalid && k < KTOT) {
          int khkw = k / CIN;
          int c    = k - khkw * CIN;
          int kh = khkw / 3;
          int kw = khkw - kh * 3;
          int ih = h + kh - 1;
          int iw = wcol + kw - 1;
          if (ih >= 0 && ih < H && iw >= 0 && iw < W)
            v = (float)inp[(size_t)c * HW + ih * W + iw];
        }
        a[e] = (_Float16)v;
      }
    }

    // prefetch next chunk's packed B block (4 KB, contiguous)
    if (chunk + 1 < NCHUNKS)
      __builtin_prefetch(wp + ((size_t)(chunk + 1) * 4 * 32) * 16, 0, 1);

    // ---- 4 B tiles (one per 16-channel group), 4 WMMAs ----
    const _Float16* wb = wp + ((size_t)chunk * 4 * 32 + lane) * 16;
    v16h b0 = *(const v16h*)(wb + 0 * 32 * 16);
    v16h b1 = *(const v16h*)(wb + 1 * 32 * 16);
    v16h b2 = *(const v16h*)(wb + 2 * 32 * 16);
    v16h b3 = *(const v16h*)(wb + 3 * 32 * 16);
    acc0 = __builtin_amdgcn_wmma_f32_16x16x32_f16(false, a, false, b0, (short)0, acc0, false, false);
    acc1 = __builtin_amdgcn_wmma_f32_16x16x32_f16(false, a, false, b1, (short)0, acc1, false, false);
    acc2 = __builtin_amdgcn_wmma_f32_16x16x32_f16(false, a, false, b2, (short)0, acc2, false, false);
    acc3 = __builtin_amdgcn_wmma_f32_16x16x32_f16(false, a, false, b3, (short)0, acc3, false, false);
  }

  // ---- store D + bias (NCHW, contiguous pixels) ----
  int n = lane & 15;
  int Mbase = (lane >> 4) << 3;
  int pbase = mtile * 16 + Mbase;
  v8f accs[4] = {acc0, acc1, acc2, acc3};
#pragma unroll
  for (int t = 0; t < 4; ++t) {
    int o = t * 16 + n;
    float bi = bias[o];
    float* outp = out + ((size_t)img * CHN + o) * HW;
#pragma unroll
    for (int r = 0; r < 8; ++r) {
      int pp = pbase + r;
      if (pp < HW) outp[pp] = accs[t][r] + bi;
    }
  }
}

// ---------------------------------------------------------------------------
// GroupNorm statistics: one block per (image, group of 16 channels).
// Conv buffer is NCHW f32.
// ---------------------------------------------------------------------------
__global__ __launch_bounds__(256)
void gn_stats_k(const float* __restrict__ conv, float* __restrict__ stats, int HW) {
  __shared__ float rs[256];
  __shared__ float rq[256];
  int img = blockIdx.x >> 2;
  int grp = blockIdx.x & 3;
  const float* base = conv + ((size_t)img * CHN + grp * 16) * HW;
  int cnt = 16 * HW;
  float s = 0.f, q = 0.f;
  for (int i = threadIdx.x; i < cnt; i += 256) {
    float v = base[i];
    s += v;
    q += v * v;
  }
  rs[threadIdx.x] = s;
  rq[threadIdx.x] = q;
  __syncthreads();
  for (int off = 128; off > 0; off >>= 1) {
    if (threadIdx.x < (unsigned)off) {
      rs[threadIdx.x] += rs[threadIdx.x + off];
      rq[threadIdx.x] += rq[threadIdx.x + off];
    }
    __syncthreads();
  }
  if (threadIdx.x == 0) {
    float inv = 1.f / (float)cnt;
    float mu  = rs[0] * inv;
    float var = rq[0] * inv - mu * mu;
    stats[blockIdx.x * 2 + 0] = mu;
    stats[blockIdx.x * 2 + 1] = rsqrtf(var + 1e-5f);
  }
}

// ---------------------------------------------------------------------------
// Fused normalize (affine) + ReLU + 2x2 maxpool (floor). Reads NCHW f32 conv.
// NHWC=true : writes [img][ho][wo][ch] f16 (channel-fastest thread index ->
//             fully coalesced stores) for the next conv's A loads.
// NHWC=false: writes [img][ch][ho][wo] (embedding flatten order).
// ---------------------------------------------------------------------------
template <typename TOUT, bool NHWC>
__global__ void gn_pool_k(const float* __restrict__ conv, const float* __restrict__ stats,
                          const float* __restrict__ g, const float* __restrict__ beta,
                          TOUT* __restrict__ out, int H, int W, int Ho, int Wo, int total) {
  int idx = blockIdx.x * blockDim.x + threadIdx.x;
  if (idx >= total) return;
  int o, ho, wo, img;
  if constexpr (NHWC) {
    o = idx & (CHN - 1);
    int r = idx >> 6;
    wo = r % Wo;  r /= Wo;
    ho = r % Ho;
    img = r / Ho;
  } else {
    wo = idx % Wo;
    int r = idx / Wo;
    ho = r % Ho;  r /= Ho;
    o  = r % CHN;
    img = r / CHN;
  }
  int grp = o >> 4;
  float mu   = stats[(img * NGRP + grp) * 2 + 0];
  float rstd = stats[(img * NGRP + grp) * 2 + 1];
  float sc = rstd * g[o];
  float sh = beta[o] - mu * sc;
  const float* base = conv + ((size_t)img * CHN + o) * H * W + (2 * ho) * W + 2 * wo;
  float a0 = fmaxf(base[0]     * sc + sh, 0.f);
  float a1 = fmaxf(base[1]     * sc + sh, 0.f);
  float a2 = fmaxf(base[W]     * sc + sh, 0.f);
  float a3 = fmaxf(base[W + 1] * sc + sh, 0.f);
  // In both layouts the destination flat offset equals idx.
  out[idx] = (TOUT)fmaxf(fmaxf(a0, a1), fmaxf(a2, a3));
}

// ---------------------------------------------------------------------------
// Class prototypes: deterministic gather (no atomics).
// ---------------------------------------------------------------------------
__global__ __launch_bounds__(256)
void proto_k(const float* __restrict__ es, const int* __restrict__ y,
             float* __restrict__ c, int S) {
  int b = blockIdx.x / NCLS;
  int k = blockIdx.x % NCLS;
  for (int d = threadIdx.x; d < DEMB; d += 256) {
    float acc = 0.f;
    for (int s = 0; s < S; ++s)
      for (int n = 0; n < NCLS; ++n)
        if (y[(b * S + s) * NCLS + n] == k)
          acc += es[(size_t)((b * S + s) * NCLS + n) * DEMB + d];
    c[(size_t)(b * NCLS + k) * DEMB + d] = acc / (float)S;
  }
}

// ---------------------------------------------------------------------------
// Euclidean distance: one block per (b, query, class), LDS tree reduction.
// ---------------------------------------------------------------------------
__global__ __launch_bounds__(256)
void dist_k(const float* __restrict__ eq, const float* __restrict__ c,
            float* __restrict__ out, int QN) {
  __shared__ float rs[256];
  int k  = blockIdx.x % NCLS;
  int t  = blockIdx.x / NCLS;
  int qn = t % QN;
  int b  = t / QN;
  const float* e  = eq + (size_t)(b * QN + qn) * DEMB;
  const float* cc = c  + (size_t)(b * NCLS + k) * DEMB;
  float s = 0.f;
  for (int d = threadIdx.x; d < DEMB; d += 256) {
    float diff = e[d] - cc[d];
    s += diff * diff;
  }
  rs[threadIdx.x] = s;
  __syncthreads();
  for (int off = 128; off > 0; off >>= 1) {
    if (threadIdx.x < (unsigned)off) rs[threadIdx.x] += rs[threadIdx.x + off];
    __syncthreads();
  }
  if (threadIdx.x == 0) out[blockIdx.x] = sqrtf(rs[0]);
}

// ---------------------------------------------------------------------------
// Host orchestration
// ---------------------------------------------------------------------------
extern "C" void kernel_launch(void* const* d_in, const int* in_sizes, int n_in,
                              void* d_out, int out_size, void* d_ws, size_t ws_size,
                              hipStream_t stream) {
  (void)in_sizes; (void)n_in; (void)out_size; (void)ws_size;

  const float* supp = (const float*)d_in[0];   // [100][3][84][84]
  const int*   ytr  = (const int*)d_in[1];     // [4][5][5]
  const float* qry  = (const float*)d_in[2];   // [300][3][84][84]
  const float *w[4], *bb[4], *gg[4], *be[4];
  for (int i = 0; i < 4; ++i) {
    w[i]  = (const float*)d_in[3 + 4 * i];
    bb[i] = (const float*)d_in[4 + 4 * i];
    gg[i] = (const float*)d_in[5 + 4 * i];
    be[i] = (const float*)d_in[6 + 4 * i];
  }

  const int B = 4, S = 5, Q = 15;
  const int Nsup = B * S * NCLS;   // 100
  const int Nqry = B * Q * NCLS;   // 300
  const int CHUNK = 10;            // divides 100 and 300; ws stays ~23 MB

  // ---- workspace carve-up (256B aligned) ----
  char* ws = (char*)d_ws;
  size_t off = 0;
  auto carve = [&](size_t bytes) -> char* {
    char* p = ws + off;
    off += (bytes + 255) & ~((size_t)255);
    return p;
  };
  const int NCH1 = 1, NCH = 18;  // K chunks: 27->1, 576->18
  _Float16* wp1 = (_Float16*)carve((size_t)NCH1 * 4 * 512 * sizeof(_Float16));
  _Float16* wp2 = (_Float16*)carve((size_t)NCH  * 4 * 512 * sizeof(_Float16));
  _Float16* wp3 = (_Float16*)carve((size_t)NCH  * 4 * 512 * sizeof(_Float16));
  _Float16* wp4 = (_Float16*)carve((size_t)NCH  * 4 * 512 * sizeof(_Float16));
  float*    convbuf = (float*)carve((size_t)CHUNK * CHN * 84 * 84 * sizeof(float));
  _Float16* actbuf  = (_Float16*)carve((size_t)CHUNK * CHN * 42 * 42 * sizeof(_Float16));
  float*    stats   = (float*)carve((size_t)CHUNK * NGRP * 2 * sizeof(float));
  float*    embeds  = (float*)carve((size_t)(Nsup + Nqry) * DEMB * sizeof(float));
  float*    protos  = (float*)carve((size_t)B * NCLS * DEMB * sizeof(float));

  // ---- pack weights into WMMA B-fragment layout ----
  {
    int t1 = NCH1 * 4 * 512;
    pack_weights_k<<<(t1 + 255) / 256, 256, 0, stream>>>(w[0], wp1, 3, NCH1);
    int t2 = NCH * 4 * 512;
    pack_weights_k<<<(t2 + 255) / 256, 256, 0, stream>>>(w[1], wp2, CHN, NCH);
    pack_weights_k<<<(t2 + 255) / 256, 256, 0, stream>>>(w[2], wp3, CHN, NCH);
    pack_weights_k<<<(t2 + 255) / 256, 256, 0, stream>>>(w[3], wp4, CHN, NCH);
  }

  auto run_embed = [&](const float* imgs, float* embase, int nimg) {
    for (int c0 = 0; c0 < nimg; c0 += CHUNK) {
      const int n = CHUNK;
      const float* lin = imgs + (size_t)c0 * 3 * 84 * 84;

      // ---- layer 1: 3x84x84 (NCHW f32) -> 64x84x84 -> pool -> NHWC f16 42x42 ----
      {
        constexpr int NMB = (84 * 84 + 63) / 64;   // 111
        conv_wmma_k<float, 3, 84, 84><<<n * NMB, 128, 0, stream>>>(lin, wp1, bb[0], convbuf);
        gn_stats_k<<<n * NGRP, 256, 0, stream>>>(convbuf, stats, 84 * 84);
        int tot = n * CHN * 42 * 42;
        gn_pool_k<_Float16, true><<<(tot + 255) / 256, 256, 0, stream>>>(
            convbuf, stats, gg[0], be[0], actbuf, 84, 84, 42, 42, tot);
      }
      // ---- layer 2: NHWC 64x42x42 -> pool -> NHWC f16 21x21 ----
      {
        constexpr int NMB = (42 * 42 + 63) / 64;   // 28
        conv_wmma_k<_Float16, 64, 42, 42><<<n * NMB, 128, 0, stream>>>(actbuf, wp2, bb[1], convbuf);
        gn_stats_k<<<n * NGRP, 256, 0, stream>>>(convbuf, stats, 42 * 42);
        int tot = n * CHN * 21 * 21;
        gn_pool_k<_Float16, true><<<(tot + 255) / 256, 256, 0, stream>>>(
            convbuf, stats, gg[1], be[1], actbuf, 42, 42, 21, 21, tot);
      }
      // ---- layer 3: NHWC 64x21x21 -> pool -> NHWC f16 10x10 ----
      {
        constexpr int NMB = (21 * 21 + 63) / 64;   // 7
        conv_wmma_k<_Float16, 64, 21, 21><<<n * NMB, 128, 0, stream>>>(actbuf, wp3, bb[2], convbuf);
        gn_stats_k<<<n * NGRP, 256, 0, stream>>>(convbuf, stats, 21 * 21);
        int tot = n * CHN * 10 * 10;
        gn_pool_k<_Float16, true><<<(tot + 255) / 256, 256, 0, stream>>>(
            convbuf, stats, gg[2], be[2], actbuf, 21, 21, 10, 10, tot);
      }
      // ---- layer 4: NHWC 64x10x10 -> pool -> NCHW f32 embeddings ----
      {
        constexpr int NMB = (10 * 10 + 63) / 64;   // 2
        conv_wmma_k<_Float16, 64, 10, 10><<<n * NMB, 128, 0, stream>>>(actbuf, wp4, bb[3], convbuf);
        gn_stats_k<<<n * NGRP, 256, 0, stream>>>(convbuf, stats, 10 * 10);
        int tot = n * CHN * 5 * 5;
        gn_pool_k<float, false><<<(tot + 255) / 256, 256, 0, stream>>>(
            convbuf, stats, gg[3], be[3], embase + (size_t)c0 * DEMB, 10, 10, 5, 5, tot);
      }
    }
  };

  run_embed(supp, embeds, Nsup);
  run_embed(qry, embeds + (size_t)Nsup * DEMB, Nqry);

  proto_k<<<B * NCLS, 256, 0, stream>>>(embeds, ytr, protos, S);
  const int QN = Q * NCLS;  // 75
  dist_k<<<B * QN * NCLS, 256, 0, stream>>>(embeds + (size_t)Nsup * DEMB, protos,
                                            (float*)d_out, QN);
}
